// MwAN_2508260901185
// MI455X (gfx1250) — matrix operand order repaired and checked by reading, not compiled, not run on hardware
//
#include <hip/hip_runtime.h>
#include <hip/hip_bf16.h>
#include <math.h>

// ---------------------------------------------------------------------------
// MwAN forward for MI455X (gfx1250, wave32, WMMA f32<=f16 16x16x32,
// async global->LDS staging for the GRU recurrent weights)
// B=4, L=256, E=128, H=128, D=256, V=32000
// ---------------------------------------------------------------------------

typedef __attribute__((ext_vector_type(16))) _Float16 v16h;
typedef __attribute__((ext_vector_type(8)))  float    v8f;

#define Bq 4
#define Lq 256
#define Eq 128
#define Hq 128
#define Dq 256   // 2*H
#define G3 384   // 3*H
#define WHH_PITCH 129   // 129 mod 64 == 1 -> conflict-free in-wave LDS reads

// ---------------- reduction helpers (wave32) ----------------
__device__ __forceinline__ float waveSum(float v) {
#pragma unroll
  for (int o = 16; o; o >>= 1) v += __shfl_xor(v, o, 32);
  return v;
}
__device__ __forceinline__ float waveMax(float v) {
#pragma unroll
  for (int o = 16; o; o >>= 1) v = fmaxf(v, __shfl_xor(v, o, 32));
  return v;
}
__device__ __forceinline__ float blockSum(float v, float* red) {
  v = waveSum(v);
  int lane = threadIdx.x & 31, w = threadIdx.x >> 5;
  if (lane == 0) red[w] = v;
  __syncthreads();
  int nw = blockDim.x >> 5;
  float r = (threadIdx.x < (unsigned)nw) ? red[threadIdx.x] : 0.f;
  r = waveSum(r);
  if (threadIdx.x == 0) red[0] = r;
  __syncthreads();
  r = red[0];
  __syncthreads();
  return r;
}
__device__ __forceinline__ float blockMax(float v, float* red) {
  v = waveMax(v);
  int lane = threadIdx.x & 31, w = threadIdx.x >> 5;
  if (lane == 0) red[w] = v;
  __syncthreads();
  int nw = blockDim.x >> 5;
  float r = (threadIdx.x < (unsigned)nw) ? red[threadIdx.x] : -1e30f;
  r = waveMax(r);
  if (threadIdx.x == 0) red[0] = r;
  __syncthreads();
  r = red[0];
  __syncthreads();
  return r;
}
__device__ __forceinline__ float sigmoidf(float x) { return 1.f / (1.f + expf(-x)); }

// ---------------- embedding gather ----------------
__global__ void k_embed(const int* __restrict__ idx, const float* __restrict__ tab,
                        float* __restrict__ out, int rows) {
  long i = (long)blockIdx.x * blockDim.x + threadIdx.x;
  if (i >= (long)rows * Eq) return;
  int r = (int)(i >> 7), e = (int)(i & 127);
  out[i] = tab[(long)idx[r] * Eq + e];
}

// ---------------- generic WMMA GEMM: Y[M,N] = X[M,K] @ W[N,K]^T (+bias) ------
// block = 128 threads (4 waves); each wave computes a 64(M) x 32(N) register
// tile (4x2 WMMA tiles, 8 accumulators): each B-pair feeds 4 WMMAs, each
// A-fragment feeds 2. Requires M%64==0, N%32==0 (grid: N/128 x M/64), K%32==0.
__global__ void k_gemm_wmma(const float* __restrict__ X, const float* __restrict__ W,
                            const float* __restrict__ bias, float* __restrict__ Y,
                            int M, int N, int K,
                            long sXb, long sWb, long sYb) {
  int bz = blockIdx.z;
  X += (long)bz * sXb; W += (long)bz * sWb; Y += (long)bz * sYb;
  int wave = threadIdx.x >> 5;
  int lane = threadIdx.x & 31;
  int mloc = lane & 15;
  int half = lane >> 4;
  int tM0 = blockIdx.y * 4;                    // 4 consecutive M-tiles
  int tN0 = (blockIdx.x * 4 + wave) * 2;       // 2 consecutive N-tiles
  if (tN0 * 16 >= N || tM0 * 16 >= M) return;
  const float* arow[4];
#pragma unroll
  for (int mt = 0; mt < 4; ++mt)
    arow[mt] = X + (long)((tM0 + mt) * 16 + mloc) * K;
  const float* brow0 = W + (long)(tN0 * 16 + mloc) * K + half * 16;
  const float* brow1 = brow0 + (long)16 * K;   // next N-tile
  int kA = half * 16;
  v8f acc[4][2] = {};
  for (int k0 = 0; k0 < K; k0 += 32) {
    v16h bf0, bf1;
#pragma unroll
    for (int j = 0; j < 16; ++j) {
      bf0[j] = (_Float16)brow0[k0 + j];
      bf1[j] = (_Float16)brow1[k0 + j];
    }
#pragma unroll
    for (int mt = 0; mt < 4; ++mt) {
      v16h af;
#pragma unroll
      for (int j = 0; j < 8; ++j) {
        af[j]     = (_Float16)arow[mt][k0 + kA + j];
        af[j + 8] = (_Float16)arow[mt][k0 + kA + 8 + j];
      }
      acc[mt][0] = __builtin_amdgcn_wmma_f32_16x16x32_f16(false, af, false, bf0,
                                                          (short)0, acc[mt][0], false, false);
      acc[mt][1] = __builtin_amdgcn_wmma_f32_16x16x32_f16(false, af, false, bf1,
                                                          (short)0, acc[mt][1], false, false);
    }
  }
#pragma unroll
  for (int mt = 0; mt < 4; ++mt) {
    int mBase = (tM0 + mt) * 16 + half * 8;
#pragma unroll
    for (int nt = 0; nt < 2; ++nt) {
      int nOut = (tN0 + nt) * 16 + mloc;
      float bv = bias ? bias[nOut] : 0.f;
#pragma unroll
      for (int r = 0; r < 8; ++r)
        Y[(long)(mBase + r) * N + nOut] = acc[mt][nt][r] + bv;
    }
  }
}

// ---------------- fused dot-attention scores (the 17 GFLOP GEMM) -------------
// score[b,c,p] = sum_h tanh( sum_d hp[b,p,d]*hc[b,c,d]*Wd[h,d] ) * Vd[h]
// one wave per (b, c, ptile); the elementwise-scaled A fragment is loaded once
// per K-step and amortized over all 8 h-tiles (8 accumulators, 8 static WMMAs);
// tanh*Vd row-reduction fused in the epilogue (no (B,L,L,H) materialization).
__global__ void k_score_dot(const float* __restrict__ hp, const float* __restrict__ hc,
                            const float* __restrict__ Wd, const float* __restrict__ Vd,
                            float* __restrict__ scores) {
  int gwave = blockIdx.x * (blockDim.x >> 5) + (threadIdx.x >> 5);
  int ptile = gwave & 15;
  int c = (gwave >> 4) & (Lq - 1);
  int b = gwave >> 12;
  int lane = threadIdx.x & 31;
  int mloc = lane & 15, half = lane >> 4;
  int p = ptile * 16 + mloc;
  const float* arow = hp + ((long)b * Lq + p) * Dq;
  const float* crow = hc + ((long)b * Lq + c) * Dq;
  int kA = half * 16;
  v8f acc[8] = {};
  for (int k0 = 0; k0 < Dq; k0 += 32) {
    v16h af;
#pragma unroll
    for (int j = 0; j < 8; ++j) {
      af[j]     = (_Float16)(arow[k0 + kA + j]      * crow[k0 + kA + j]);
      af[j + 8] = (_Float16)(arow[k0 + kA + 8 + j] * crow[k0 + kA + 8 + j]);
    }
#pragma unroll
    for (int ht = 0; ht < 8; ++ht) {
      const float* brow = Wd + (long)(ht * 16 + mloc) * Dq + half * 16 + k0;
      v16h bf;
#pragma unroll
      for (int j = 0; j < 16; ++j) bf[j] = (_Float16)brow[j];
      acc[ht] = __builtin_amdgcn_wmma_f32_16x16x32_f16(false, af, false, bf,
                                                       (short)0, acc[ht], false, false);
    }
  }
  float rowsum[8];
#pragma unroll
  for (int r = 0; r < 8; ++r) rowsum[r] = 0.f;
#pragma unroll
  for (int ht = 0; ht < 8; ++ht) {
    float vd = Vd[ht * 16 + mloc];            // this lane's output column h
#pragma unroll
    for (int r = 0; r < 8; ++r) rowsum[r] += tanhf(acc[ht][r]) * vd;
  }
  // sum across the 16 columns (lanes with same half)
#pragma unroll
  for (int r = 0; r < 8; ++r) {
    float v = rowsum[r];
    v += __shfl_xor(v, 1, 32); v += __shfl_xor(v, 2, 32);
    v += __shfl_xor(v, 4, 32); v += __shfl_xor(v, 8, 32);
    rowsum[r] = v;
  }
  if (mloc == 0) {
    int pBase = ptile * 16 + half * 8;
#pragma unroll
    for (int r = 0; r < 8; ++r)
      scores[((long)b * Lq + c) * Lq + pBase + r] = rowsum[r];
  }
}

// ---------------- concat/minus attention scores (factored form) --------------
// score[b,c,p] = sum_h tanh(Ap[b,p,h] + sign*Ac[b,c,h]) * V[h]
// one wave per p-row (8 waves/block, no block barriers in the hot loop)
__global__ void k_score_addtanh(const float* __restrict__ Ap, const float* __restrict__ Ac,
                                const float* __restrict__ V, float* __restrict__ scores,
                                float sign) {
  int c = blockIdx.x, b = blockIdx.y;
  int wave = threadIdx.x >> 5, lane = threadIdx.x & 31;   // 256 threads
  const float4 ac4 = ((const float4*)(Ac + ((long)b * Lq + c) * Hq))[lane];
  const float4 v4  = ((const float4*)V)[lane];
  float acx = ac4.x * sign, acy = ac4.y * sign, acz = ac4.z * sign, acw = ac4.w * sign;
  for (int p = wave; p < Lq; p += 8) {
    const float4 a4 = ((const float4*)(Ap + ((long)b * Lq + p) * Hq))[lane];
    float s = tanhf(a4.x + acx) * v4.x + tanhf(a4.y + acy) * v4.y +
              tanhf(a4.z + acz) * v4.z + tanhf(a4.w + acw) * v4.w;
    s = waveSum(s);
    if (lane == 0) scores[((long)b * Lq + c) * Lq + p] = s;
  }
}

// ---------------- softmax over p + apply: rep[b,c,:] = softmax(scores)@src ---
// grid (Lc, B), block 256 (= L = D)
__global__ void k_attn_apply(const float* __restrict__ scores, const float* __restrict__ src,
                             float* __restrict__ rep, int Lc) {
  int c = blockIdx.x, b = blockIdx.y;
  int t = threadIdx.x;
  __shared__ float a[Lq];
  __shared__ float red[8];
  const float* srow = scores + ((long)b * Lc + c) * Lq;
  float v = srow[t];
  float m = blockMax(v, red);
  float e = expf(v - m);
  float s = blockSum(e, red);
  a[t] = e / s;
  __syncthreads();
  float acc = 0.f;
  for (int p = 0; p < Lq; ++p)
    acc += a[p] * src[((long)b * Lq + p) * Dq + t];
  rep[((long)b * Lc + c) * Dq + t] = acc;
}

// ---------------- GRU scan (one block per direction, all 4 batches) ----------
// block = 384 threads; h kept in LDS; Whh staged once into LDS (padded pitch)
// via the CDNA5 async global->LDS path, then re-read 256x from LDS.
__global__ void k_gru_scan(const float* __restrict__ xgF, const float* __restrict__ xgB,
                           const float* __restrict__ WhhF, const float* __restrict__ bhhF,
                           const float* __restrict__ WhhB, const float* __restrict__ bhhB,
                           float* __restrict__ out) {
  extern __shared__ float smem[];
  float* whh_s = smem;                               // G3 rows x WHH_PITCH
  float* h_lds = smem + G3 * WHH_PITCH;              // 4 x 128
  float* hg    = h_lds + Bq * Hq;                    // 4 x 384
  int dir = blockIdx.x;
  const float* xg  = dir ? xgB  : xgF;
  const float* Whh = dir ? WhhB : WhhF;
  const float* bhh = dir ? bhhB : bhhF;
  int t = threadIdx.x;
  // ---- async-stage Whh (3H x H) into LDS with conflict-free padding ----
  for (int i = t; i < G3 * Hq; i += G3) {
    int row = i >> 7, k = i & 127;
    unsigned loff = (unsigned)(row * WHH_PITCH + k) * 4u;
    const float* gp = Whh + i;
    asm volatile("global_load_async_to_lds_b32 %0, %1, off"
                 :: "v"(loff), "v"(gp) : "memory");
  }
  if (t < Hq)
    for (int b = 0; b < Bq; ++b) h_lds[b * Hq + t] = 0.f;
  float bia = bhh[t];                                // hoisted: invariant per step
  asm volatile("s_wait_asynccnt 0x0" ::: "memory");
  __syncthreads();
  for (int s = 0; s < Lq; ++s) {
    int time = dir ? (Lq - 1 - s) : s;
    // phase 1: hg = h @ Whh^T + bhh  (384 dots x 4 batches, Whh from LDS)
    {
      const float* wr = whh_s + t * WHH_PITCH;
      float a0 = bia, a1 = bia, a2 = bia, a3 = bia;
      for (int k = 0; k < Hq; ++k) {
        float w = wr[k];
        a0 += w * h_lds[0 * Hq + k];
        a1 += w * h_lds[1 * Hq + k];
        a2 += w * h_lds[2 * Hq + k];
        a3 += w * h_lds[3 * Hq + k];
      }
      hg[0 * G3 + t] = a0; hg[1 * G3 + t] = a1;
      hg[2 * G3 + t] = a2; hg[3 * G3 + t] = a3;
    }
    __syncthreads();
    float hnew[Bq];
    if (t < Hq) {
      for (int b = 0; b < Bq; ++b) {
        long row = ((long)b * Lq + time) * G3;
        float xr = xg[row + t], xz = xg[row + Hq + t], xn = xg[row + 2 * Hq + t];
        float r = sigmoidf(xr + hg[b * G3 + t]);
        float z = sigmoidf(xz + hg[b * G3 + Hq + t]);
        float n = tanhf(xn + r * hg[b * G3 + 2 * Hq + t]);
        hnew[b] = (1.f - z) * n + z * h_lds[b * Hq + t];
        out[((long)b * Lq + time) * Dq + dir * Hq + t] = hnew[b];
      }
    }
    __syncthreads();
    if (t < Hq)
      for (int b = 0; b < Bq; ++b) h_lds[b * Hq + t] = hnew[b];
    __syncthreads();
  }
}

// ---------------- gate: xj=[rep,hp]; gj=sigmoid(xj.Wg); gated=gj*xj ----------
__global__ void k_gate(const float* __restrict__ rep, const float* __restrict__ hp,
                       const float* __restrict__ Wg, float* __restrict__ gated) {
  int row = blockIdx.x;     // b*L + l
  int d = threadIdx.x;      // 256
  __shared__ float red[8];
  float rv = rep[(long)row * Dq + d];
  float hv = hp[(long)row * Dq + d];
  float part = rv * Wg[d] + hv * Wg[Dq + d];
  float s = blockSum(part, red);
  float g = sigmoidf(s);
  gated[(long)row * (2 * Dq) + d] = g * rv;
  gated[(long)row * (2 * Dq) + Dq + d] = g * hv;
}

// ---------------- x_cat_v = (x_cat @ Wx^T) + vx ------------------------------
__global__ void k_xcv(const float* __restrict__ a0, const float* __restrict__ a1,
                      const float* __restrict__ a2, const float* __restrict__ a3,
                      const float* __restrict__ Wx, const float* __restrict__ vx,
                      float* __restrict__ xcv) {
  int row = blockIdx.x;     // b*L + l
  int l = row & (Lq - 1);
  int d = threadIdx.x;      // 256
  __shared__ float red[8];
  float w = Wx[d];
  const float* arr[4] = {a0, a1, a2, a3};
#pragma unroll
  for (int a = 0; a < 4; ++a) {
    float s = blockSum(arr[a][(long)row * Dq + d] * w, red);
    if (d == 0) xcv[(long)row * 4 + a] = s + vx[l * 4 + a];
  }
}

// ---------------- sx = xcv^T @ Vx ; cross_attn = softmax over batch ----------
__global__ void k_cross(const float* __restrict__ xcv, const float* __restrict__ Vx,
                        float* __restrict__ attn) {
  __shared__ float sx[16];
  int t = threadIdx.x;      // 32 threads
  if (t < 16) {
    int b = t >> 2, a = t & 3;
    float s = 0.f;
    for (int l = 0; l < Lq; ++l) s += xcv[((long)b * Lq + l) * 4 + a] * Vx[l];
    sx[t] = s;
  }
  __syncthreads();
  if (t < 4) {              // a = t ; softmax over b (reference: axis=0)
    float m = -1e30f;
    for (int b = 0; b < Bq; ++b) m = fmaxf(m, sx[b * 4 + t]);
    float e[Bq]; float sum = 0.f;
    for (int b = 0; b < Bq; ++b) { e[b] = expf(sx[b * 4 + t] - m); sum += e[b]; }
    for (int b = 0; b < Bq; ++b) attn[b * 4 + t] = e[b] / sum;
  }
}

// ---------------- agg[b,l,d] = sum_a x_cat_a * attn[b,a] --------------------
__global__ void k_mixagg(const float* __restrict__ a0, const float* __restrict__ a1,
                         const float* __restrict__ a2, const float* __restrict__ a3,
                         const float* __restrict__ attn, float* __restrict__ out) {
  long i = (long)blockIdx.x * blockDim.x + threadIdx.x;   // B*L*D
  if (i >= (long)Bq * Lq * Dq) return;
  int b = (int)(i >> 16);   // L*D = 65536
  out[i] = a0[i] * attn[b * 4 + 0] + a1[i] * attn[b * 4 + 1] +
           a2[i] * attn[b * 4 + 2] + a3[i] * attn[b * 4 + 3];
}

// ---------------- sp[row] = sum_h tanh(X[row,h]) * V[h] ----------------------
__global__ void k_rowtanh_dot(const float* __restrict__ X, const float* __restrict__ V,
                              float* __restrict__ out) {
  int row = blockIdx.x;
  int h = threadIdx.x;      // 128
  __shared__ float red[8];
  float s = blockSum(tanhf(X[(long)row * Hq + h]) * V[h], red);
  if (h == 0) out[row] = s;
}

// ---------------- rpW2[b,h] = sum_d rp[b,d]*W2[h,d] --------------------------
__global__ void k_rpW2(const float* __restrict__ rp, const float* __restrict__ W2,
                       float* __restrict__ out) {
  int i = blockIdx.x * blockDim.x + threadIdx.x;   // B*H = 512
  if (i >= Bq * Hq) return;
  int b = i >> 7, h = i & 127;
  float s = 0.f;
  for (int d = 0; d < Dq; ++d) s += rp[b * Dq + d] * W2[h * Dq + d];
  out[i] = s;
}

// ---------------- s2[row] = sum_h (A1[row,h]+rpW2[b,h]) * V[h] ---------------
__global__ void k_s2(const float* __restrict__ A1, const float* __restrict__ rpW2,
                     const float* __restrict__ V, float* __restrict__ s2) {
  int row = blockIdx.x;     // b*L + l
  int b = row >> 8;
  int h = threadIdx.x;      // 128
  __shared__ float red[8];
  float s = blockSum((A1[(long)row * Hq + h] + rpW2[b * Hq + h]) * V[h], red);
  if (h == 0) s2[row] = s;
}

// ---------------- out = rc @ out_W^T + out_b ---------------------------------
__global__ void k_final(const float* __restrict__ rc, const float* __restrict__ W,
                        const float* __restrict__ bias, float* __restrict__ out) {
  int t = threadIdx.x;
  if (t < Bq * 2) {
    int b = t >> 1, j = t & 1;
    float s = bias[j];
    for (int d = 0; d < Dq; ++d) s += rc[b * Dq + d] * W[j * Dq + d];
    out[b * 2 + j] = s;
  }
}

// ===========================================================================
extern "C" void kernel_launch(void* const* d_in, const int* in_sizes, int n_in,
                              void* d_out, int out_size, void* d_ws, size_t ws_size,
                              hipStream_t stream) {
  (void)in_sizes; (void)n_in; (void)out_size; (void)ws_size;
  auto f = [&](int i) { return (const float*)d_in[i]; };
  const int* text_a = (const int*)d_in[0];
  const int* text_b = (const int*)d_in[1];
  const float* emb = f(2);
  // GRU param bases (tuple order Wih, Whh, bih, bhh; fwd then bwd)
  const int P_GRU = 3, C_GRU = 11, I_GRU = 19, M_GRU = 27;
  const float* Wc1 = f(35); const float* Wc2 = f(36); const float* Vc = f(37);
  const float* Wb  = f(38); const float* Wd  = f(39); const float* Vd = f(40);
  const float* Wm  = f(41); const float* Vm  = f(42); const float* Wg = f(43);
  const float* Wx  = f(44); const float* vx  = f(45); const float* Vx = f(46);
  const float* Wp  = f(47); const float* Vp  = f(48); const float* W1 = f(49);
  const float* W2p = f(50); const float* Vv  = f(51);
  const float* outW = f(52); const float* outb = f(53);

  // ---- workspace carve (floats) ----
  float* w = (float*)d_ws;
  auto carve = [&](size_t n) { float* p = w; w += n; return p; };
  float* embA  = carve(Bq * Lq * Eq);       // 131072
  float* embB  = carve(Bq * Lq * Eq);
  float* xgF   = carve(Bq * Lq * G3);       // 393216
  float* xgB   = carve(Bq * Lq * G3);
  float* hp    = carve(Bq * Lq * Dq);       // 262144
  float* hc    = carve(Bq * Lq * Dq);
  float* Ap    = carve(Bq * Lq * Hq);       // 131072
  float* Ac    = carve(Bq * Lq * Hq);
  float* hpWb  = carve(Bq * Lq * Dq);
  float* scores= carve(Bq * Lq * Lq);       // 262144
  float* rep   = carve(Bq * Lq * Dq);
  float* gated = carve(Bq * Lq * 2 * Dq);   // 524288
  float* agg[4];
  for (int i = 0; i < 4; ++i) agg[i] = carve(Bq * Lq * Dq);
  float* xcv   = carve(Bq * Lq * 4);
  float* attn  = carve(16);
  float* aggmix= carve(Bq * Lq * Dq);
  float* aggrep= carve(Bq * Lq * Dq);
  float* PA    = carve(Bq * Lq * Hq);       // shared: hp@Wp^T then aggrep@W1^T
  float* sp    = carve(Bq * Lq);
  float* rp    = carve(Bq * Dq);
  float* rpW2  = carve(Bq * Hq);
  float* s2    = carve(Bq * Lq);
  float* rc    = carve(Bq * Dq);

  const size_t gruLds = (size_t)(G3 * WHH_PITCH + Bq * Hq + Bq * G3) * sizeof(float);

  auto gemm = [&](const float* X, const float* Wt, const float* bias, float* Y,
                  int M, int N, int K, int batch, long sx, long sw, long sy) {
    dim3 g(N / 128, M / 64, batch);
    k_gemm_wmma<<<g, 128, 0, stream>>>(X, Wt, bias, Y, M, N, K, sx, sw, sy);
  };
  auto bigru = [&](const float* x, int Kin, int base, float* out) {
    gemm(x, f(base + 0), f(base + 2), xgF, Bq * Lq, G3, Kin, 1, 0, 0, 0);
    gemm(x, f(base + 4), f(base + 6), xgB, Bq * Lq, G3, Kin, 1, 0, 0, 0);
    k_gru_scan<<<2, G3, gruLds, stream>>>(xgF, xgB, f(base + 1), f(base + 3),
                                          f(base + 5), f(base + 7), out);
  };
  auto inside = [&](float* rep_, float* aggK) {
    k_gate<<<Bq * Lq, Dq, 0, stream>>>(rep_, hp, Wg, gated);
    bigru(gated, 2 * Dq, I_GRU, aggK);
  };
  dim3 gBC(Lq, Bq);

  // ---- 1. embeddings + encoder bi-GRUs ----
  k_embed<<<(Bq * Lq * Eq) / 256, 256, 0, stream>>>(text_a, emb, embA, Bq * Lq);
  k_embed<<<(Bq * Lq * Eq) / 256, 256, 0, stream>>>(text_b, emb, embB, Bq * Lq);
  bigru(embA, Eq, P_GRU, hp);
  bigru(embB, Eq, C_GRU, hc);

  // ---- 2. concat attention ----
  gemm(hp, Wc1, nullptr, Ap, Bq * Lq, Hq, Dq, 1, 0, 0, 0);
  gemm(hc, Wc2, nullptr, Ac, Bq * Lq, Hq, Dq, 1, 0, 0, 0);
  k_score_addtanh<<<gBC, 256, 0, stream>>>(Ap, Ac, Vc, scores, 1.f);
  k_attn_apply<<<gBC, Lq, 0, stream>>>(scores, hp, rep, Lq);
  inside(rep, agg[0]);

  // ---- 3. bilinear attention ----
  gemm(hp, Wb, nullptr, hpWb, Bq * Lq, Dq, Dq, 1, 0, 0, 0);
  gemm(hc, hpWb, nullptr, scores, Lq, Lq, Dq, Bq,
       (long)Lq * Dq, (long)Lq * Dq, (long)Lq * Lq);   // batched X@W^T per b
  k_attn_apply<<<gBC, Lq, 0, stream>>>(scores, hp, rep, Lq);
  inside(rep, agg[1]);

  // ---- 4. dot attention (fused elementwise-scaled WMMA GEMM) ----
  k_score_dot<<<(Bq * Lq * (Lq / 16)) / 8, 256, 0, stream>>>(hp, hc, Wd, Vd, scores);
  k_attn_apply<<<gBC, Lq, 0, stream>>>(scores, hp, rep, Lq);
  inside(rep, agg[2]);

  // ---- 5. minus attention (algebraically factored) ----
  gemm(hp, Wm, nullptr, Ap, Bq * Lq, Hq, Dq, 1, 0, 0, 0);
  gemm(hc, Wm, nullptr, Ac, Bq * Lq, Hq, Dq, 1, 0, 0, 0);
  k_score_addtanh<<<gBC, 256, 0, stream>>>(Ap, Ac, Vm, scores, -1.f);
  k_attn_apply<<<gBC, Lq, 0, stream>>>(scores, hp, rep, Lq);
  inside(rep, agg[3]);

  // ---- 6. cross-view mixing + mix bi-GRU ----
  k_xcv<<<Bq * Lq, Dq, 0, stream>>>(agg[0], agg[1], agg[2], agg[3], Wx, vx, xcv);
  k_cross<<<1, 32, 0, stream>>>(xcv, Vx, attn);
  k_mixagg<<<(Bq * Lq * Dq) / 256, 256, 0, stream>>>(agg[0], agg[1], agg[2], agg[3],
                                                     attn, aggmix);
  bigru(aggmix, Dq, M_GRU, aggrep);

  // ---- 7. rp = softmax(Vp . tanh(hp@Wp^T)) @ hp ----
  gemm(hp, Wp, nullptr, PA, Bq * Lq, Hq, Dq, 1, 0, 0, 0);
  k_rowtanh_dot<<<Bq * Lq, Hq, 0, stream>>>(PA, Vp, sp);
  k_attn_apply<<<dim3(1, Bq), Lq, 0, stream>>>(sp, hp, rp, 1);

  // ---- 8. head ----
  gemm(aggrep, W1, nullptr, PA, Bq * Lq, Hq, Dq, 1, 0, 0, 0);
  k_rpW2<<<2, 256, 0, stream>>>(rp, W2p, rpW2);
  k_s2<<<Bq * Lq, Hq, 0, stream>>>(PA, rpW2, Vv, s2);
  k_attn_apply<<<dim3(1, Bq), Lq, 0, stream>>>(s2, aggrep, rc, 1);
  k_final<<<1, 32, 0, stream>>>(rc, outW, outb, (float*)d_out);
}
